// QuantizedLinear_4964982194273
// MI455X (gfx1250) — matrix operand order repaired and checked by reading, not compiled, not run on hardware
//
#include <hip/hip_runtime.h>

typedef __attribute__((ext_vector_type(16))) _Float16 v16h;
typedef __attribute__((ext_vector_type(8)))  _Float16 v8h;
typedef __attribute__((ext_vector_type(4)))  _Float16 v4h;
typedef __attribute__((ext_vector_type(2)))  _Float16 v2h;
typedef __attribute__((ext_vector_type(8)))  float    v8f;
typedef __attribute__((ext_vector_type(4)))  float    v4f;
typedef __attribute__((ext_vector_type(4)))  unsigned int v4u;  // clang vector (NT-load legal)

#define IN_F   8192
#define OUT_F  8192
#define M_TOK  32
#define KSPLIT 4
#define BLOCK  (KSPLIT * 32)   // 4 wave32 per block

// ---------------------------------------------------------------------------
// Pre-kernel: X fp32 -> f16 once (512 KB in d_ws).
// K-permutation: within every aligned group of 4 k's we store order
// (k0, k2, k1, k3). The same permutation falls out of the magic-number int8
// dequant on the B side, and GEMM is invariant under a consistent K-perm.
// ---------------------------------------------------------------------------
__global__ __launch_bounds__(256)
void cvt_x_f16(const float* __restrict__ x, _Float16* __restrict__ xh, int n4) {
  int i = blockIdx.x * blockDim.x + threadIdx.x;
  for (; i < n4; i += gridDim.x * blockDim.x) {
    v4f v = ((const v4f*)x)[i];
    v4h h;
    h.x = (_Float16)v.x;   // pos 0 <- k0
    h.y = (_Float16)v.z;   // pos 1 <- k2
    h.z = (_Float16)v.y;   // pos 2 <- k1
    h.w = (_Float16)v.w;   // pos 3 <- k3
    ((v4h*)xh)[i] = h;
  }
}

// Exact int8 -> f16 via magic-number: (0x6400 | (b^0x80)) as f16 == 1024+(b+128)
// Each call converts one packed u16 pair with a single v_pk_add_f16.
static __device__ __forceinline__ v2h pk_i8_to_f16(unsigned int u, v2h nbias) {
  return __builtin_bit_cast(v2h, u) + nbias;   // v_pk_add_f16
}

// ---------------------------------------------------------------------------
// Main kernel: one N=16 output-column tile per block, split-K=4 across the
// 4 waves, LDS reduction (deterministic), scale applied in epilogue.
// ---------------------------------------------------------------------------
__global__ __launch_bounds__(BLOCK)
void qlin_wmma_f16(const _Float16* __restrict__ xh,      // [32, 8192] f16 (K-permuted)
                   const signed char* __restrict__ wq,   // [8192, 8192] int8 [o][i]
                   const float* __restrict__ scale,      // [8192]
                   float* __restrict__ out) {            // [32, 8192] fp32
  __shared__ float red[KSPLIT][M_TOK * 16];

  const int lane = threadIdx.x & 31;
  const int wave = threadIdx.x >> 5;
  const int row  = lane & 15;        // N index in tile (B/C/D), M index (A)
  const int hi   = lane >> 4;        // half-wave selector
  const int n0   = blockIdx.x * 16;  // output-column tile base

  const int kBeg = wave * (IN_F / KSPLIT);
  const int kEnd = kBeg + (IN_F / KSPLIT);

  const v2h nbias = {(_Float16)(-1152.0f), (_Float16)(-1152.0f)};

  v8f acc0 = {};   // M = 0..15
  v8f acc1 = {};   // M = 16..31

  const signed char* wrow  = wq + (size_t)(n0 + row) * IN_F;
  const _Float16*    arow0 = xh + (size_t)row        * IN_F;
  const _Float16*    arow1 = xh + (size_t)(row + 16) * IN_F;

  // Unroll 8: enough independent dequant work to fill the 4 co-exec slots of
  // the WMMA->VALU WAR hazard, and ~2x the outstanding NT weight loads per
  // wave (Little's-law headroom for the 23.3 TB/s stream).
#pragma unroll 8
  for (int k0 = kBeg; k0 < kEnd; k0 += 32) {
    // ---- B tile: Wt[k0..k0+31][n0..n0+15] as f16 --------------------------
    // B in this code's contiguous-k order (matches A): lane = column N; lanes
    // 0-15 hold K 0..15, lanes 16-31 hold K 16..31. One NT b128 per lane
    // (weights are stream-once: keep them out of L2's hot set).
    const v4u wv = __builtin_nontemporal_load((const v4u*)(wrow + k0 + hi * 16));
    v16h b;
#pragma unroll
    for (int j = 0; j < 4; ++j) {
      const unsigned lo = ( wv[j]       & 0x00FF00FFu) ^ 0x64806480u; // b0,b2
      const unsigned hj = ((wv[j] >> 8) & 0x00FF00FFu) ^ 0x64806480u; // b1,b3
      const v2h p02 = pk_i8_to_f16(lo, nbias);
      const v2h p13 = pk_i8_to_f16(hj, nbias);
      b[4*j + 0] = p02.x;   // k+0
      b[4*j + 1] = p02.y;   // k+2   (group perm 0,2,1,3 — matched in Xh)
      b[4*j + 2] = p13.x;   // k+1
      b[4*j + 3] = p13.y;   // k+3
    }

    // ---- A tiles: contiguous-k 16-bit A 16x32 order (matches B) ------------------------
    // lanes 0-15: elems 0..15 = K k0+0..15
    // lanes16-31: elems 0..15 = K k0+16..31
    const v8h a0lo = *(const v8h*)(arow0 + k0 + hi * 16);
    const v8h a0hi = *(const v8h*)(arow0 + k0 + 8 + hi * 16);
    const v8h a1lo = *(const v8h*)(arow1 + k0 + hi * 16);
    const v8h a1hi = *(const v8h*)(arow1 + k0 + 8 + hi * 16);
    v16h a0, a1;
#pragma unroll
    for (int e = 0; e < 8; ++e) {
      a0[e] = a0lo[e]; a0[e + 8] = a0hi[e];
      a1[e] = a1lo[e]; a1[e + 8] = a1hi[e];
    }

    // D = A x B + C, fp32 accumulate
    acc0 = __builtin_amdgcn_wmma_f32_16x16x32_f16(false, a0, false, b,
                                                  (short)0, acc0, false, false);
    acc1 = __builtin_amdgcn_wmma_f32_16x16x32_f16(false, a1, false, b,
                                                  (short)0, acc1, false, false);
  }

  // ---- deposit partials: C/D layout -> [M][N] in LDS ----------------------
  // VGPR r holds M = hi*8 + r for lane's N = row.
#pragma unroll
  for (int r = 0; r < 8; ++r) {
    red[wave][(hi * 8 + r) * 16 + row]      = acc0[r];
    red[wave][(16 + hi * 8 + r) * 16 + row] = acc1[r];
  }
  __syncthreads();

  // ---- split-K reduction + per-channel scale + store ----------------------
  for (int idx = threadIdx.x; idx < M_TOK * 16; idx += BLOCK) {
    float s = 0.f;
#pragma unroll
    for (int w = 0; w < KSPLIT; ++w) s += red[w][idx];
    const int m = idx >> 4;
    const int c = idx & 15;
    out[(size_t)m * OUT_F + n0 + c] = s * scale[n0 + c];
  }
}

extern "C" void kernel_launch(void* const* d_in, const int* in_sizes, int n_in,
                              void* d_out, int out_size, void* d_ws, size_t ws_size,
                              hipStream_t stream) {
  const float*       x     = (const float*)d_in[0];
  const signed char* wq    = (const signed char*)d_in[1];  // int8 per reference
  const float*       scale = (const float*)d_in[2];
  float*             out   = (float*)d_out;
  _Float16*          xh    = (_Float16*)d_ws;              // 512 KB scratch

  cvt_x_f16<<<64, 256, 0, stream>>>(x, xh, M_TOK * IN_F / 4);
  qlin_wmma_f16<<<OUT_F / 16, BLOCK, 0, stream>>>(xh, wq, scale, out);
}
